// AttDecoder_14851996910065
// MI455X (gfx1250) — matrix operand order repaired and checked by reading, not compiled, not run on hardware
//
#include <hip/hip_runtime.h>
#include <math.h>

// ---------------------------------------------------------------------------
// CDNA5 / gfx1250 implementation of the attention decoder (CAN-style).
// wave32, WMMA f16->f32 (v_wmma_f32_16x16x32_f16) for all GEMM-shaped work.
// ---------------------------------------------------------------------------

typedef __attribute__((ext_vector_type(16))) _Float16 v16h;
typedef __attribute__((ext_vector_type(8)))  float    v8f;

#define B_    16
#define T_    36
#define C_    684
#define HID_  256
#define ATT_  512
#define WN_   111
#define CNTN_ 111
#define H_    16
#define W_    32
#define HW_   512   // H_*W_

// ---------------------------------------------------------------------------
// Generic tiled WMMA GEMM:  D[m,n] = act( sum_k A[m,k]*B[n,k] + bias[n] + src )
//   A element (m,k) at A[bz*batA + m*rsA + k*csA]
//   B element (n,k) at Bw[n*rsB + k*csB]            (i.e. out = A @ W^T)
//   D element (m,n) at D[bz*batD + m*rsD + n*csD]
//   src optional additive term src[(srcMmod? m%srcMmod : m)*rsS + n]
//
// One wave (32 lanes) per 16x16 tile. OOB M/N rows are only address-clamped
// (they feed exclusively unstored C rows/cols); only the K tail zero-fills.
// Main loop: branch-free batched loads -> one wait -> cvt -> wmma.
// VGPR operand maps per CDNA5 ISA 7.12.2:
//   A: k = j + (j&8) + 8*hi ; B: k = j + (j&8) + 8*hi ; C/D: m = v + 8*hi, n = lane&15.
// ---------------------------------------------------------------------------
__global__ __launch_bounds__(32) void wmma_gemm_kernel(
    const float* __restrict__ A, int rsA, int csA, int batA,
    const float* __restrict__ Bw, int rsB, int csB,
    const float* __restrict__ bias,
    const float* __restrict__ src, int rsS, int srcMmod,
    float* __restrict__ D, int rsD, int csD, int batD,
    int M, int N, int K, int act)
{
    const int lane = threadIdx.x;           // 0..31
    const int hi   = lane >> 4;             // 0 or 1
    const int r16  = lane & 15;
    const int bn   = blockIdx.x;
    const int bm   = blockIdx.y;
    const int bz   = blockIdx.z;

    const float* Ab = A + (size_t)bz * (size_t)batA;
    float*       Db = D + (size_t)bz * (size_t)batD;

    const int am  = bm * 16 + r16;          // A row this lane feeds
    const int bnr = bn * 16 + r16;          // B row (output col) this lane feeds
    // clamp OOB rows (their products land in unstored C rows/cols only)
    const float* Ap = Ab + (size_t)(am  < M ? am  : M - 1) * (size_t)rsA;
    const float* Bp = Bw + (size_t)(bnr < N ? bnr : N - 1) * (size_t)rsB;

    // loop-invariant per-element K offsets (depend on lane half hi)
    int offA[16], offB[16];
#pragma unroll
    for (int j = 0; j < 16; ++j) {
        offA[j] = (j + (j & 8) + hi * 8) * csA;
        offB[j] = (j + (j & 8) + hi * 8) * csB;
    }

    v8f c = {};
    const int kfull = K & ~31;
    int kb = 0;
    for (; kb < kfull; kb += 32) {
        float fa[16], fb[16];
#pragma unroll
        for (int j = 0; j < 16; ++j) {      // batched, branch-free loads
            fa[j] = Ap[offA[j]];
            fb[j] = Bp[offB[j]];
        }
        v16h a, b;
#pragma unroll
        for (int j = 0; j < 16; ++j) {
            a[j] = (_Float16)fa[j];
            b[j] = (_Float16)fb[j];
        }
        c = __builtin_amdgcn_wmma_f32_16x16x32_f16(false, a, false, b,
                                                   (short)0, c, false, false);
        Ap += 32 * csA;
        Bp += 32 * csB;
    }
    if (kb < K) {                           // K tail: clamp addresses, zero OOB
        float fa[16], fb[16];
#pragma unroll
        for (int j = 0; j < 16; ++j) {
            const int kA = kb + j + (j & 8) + hi * 8;
            const int kB = kb + j + (j & 8) + hi * 8;
            const int ra = (kA < K ? kA : K - 1) - kb;
            const int rb = (kB < K ? kB : K - 1) - kb;
            const float va = Ap[ra * csA];  // unconditional (clamped-safe)
            const float vb = Bp[rb * csB];
            fa[j] = (kA < K) ? va : 0.0f;
            fb[j] = (kB < K) ? vb : 0.0f;
        }
        v16h a, b;
#pragma unroll
        for (int j = 0; j < 16; ++j) {
            a[j] = (_Float16)fa[j];
            b[j] = (_Float16)fb[j];
        }
        c = __builtin_amdgcn_wmma_f32_16x16x32_f16(false, a, false, b,
                                                   (short)0, c, false, false);
    }

    // Epilogue: C/D layout -> global
    const int nOut = bn * 16 + r16;
    if (nOut < N) {
        const float bv = bias ? bias[nOut] : 0.0f;
#pragma unroll
        for (int v = 0; v < 8; ++v) {
            const int mOut = bm * 16 + v + hi * 8;
            if (mOut < M) {
                float val = c[v] + bv;
                if (src) {
                    const int sm = srcMmod ? (mOut % srcMmod) : mOut;
                    val += src[(size_t)sm * rsS + nOut];
                }
                if (act) val = tanhf(val);
                Db[(size_t)mOut * rsD + (size_t)nOut * csD] = val;
            }
        }
    }
}

// ---------------------------------------------------------------------------
// mask2d, normalized cumsums for positional embedding, avg-pool weights
// one block per batch, 512 threads (h,w)
// ---------------------------------------------------------------------------
__global__ void maskcum_kernel(const float* __restrict__ images_mask,
                               float* __restrict__ mask2d,
                               float* __restrict__ YN, float* __restrict__ XN,
                               float* __restrict__ w0)
{
    const int b = blockIdx.x, tid = threadIdx.x;
    const int h = tid >> 5, w = tid & 31;
    __shared__ float sm[HW_];
    __shared__ float red[HW_];
    // images_mask: (B,1,256,512) subsampled by RATIO=16
    const float m = images_mask[(size_t)b * 256 * 512 + (size_t)(h * 16) * 512 + (w * 16)];
    sm[tid] = m;
    __syncthreads();
    float yc = 0.f, yden = 0.f;
    for (int hh = 0; hh < H_; ++hh) { float v = sm[hh * 32 + w]; if (hh <= h) yc += v; yden += v; }
    float xc = 0.f, xden = 0.f;
    for (int ww = 0; ww < W_; ++ww) { float v = sm[h * 32 + ww]; if (ww <= w) xc += v; xden += v; }
    const float TWO_PI = 6.283185307179586f;
    YN[b * HW_ + tid] = yc / (yden + 1e-6f) * TWO_PI;
    XN[b * HW_ + tid] = xc / (xden + 1e-6f) * TWO_PI;
    mask2d[b * HW_ + tid] = m;
    red[tid] = m;
    __syncthreads();
    for (int s = 256; s > 0; s >>= 1) { if (tid < s) red[tid] += red[tid + s]; __syncthreads(); }
    w0[b * HW_ + tid] = m / red[0];    // weights for masked average pooling
}

// ---------------------------------------------------------------------------
// Add sinusoidal positional embedding into trans_hwc[b,hw,a]
// grid (hw, b), block 512 (a)
// ---------------------------------------------------------------------------
__global__ void posadd_kernel(const float* __restrict__ YN, const float* __restrict__ XN,
                              float* __restrict__ TR)
{
    const int hw = blockIdx.x, b = blockIdx.y, a = threadIdx.x;
    const float v = (a < 256) ? YN[b * HW_ + hw] : XN[b * HW_ + hw];
    const int c = a & 255;
    const int i = c >> 1;
    const float ang = v / powf(10000.0f, (float)i * (1.0f / 128.0f));
    const float p = (c & 1) ? cosf(ang) : sinf(ang);
    TR[((size_t)b * HW_ + hw) * ATT_ + a] += p;
}

// ---------------------------------------------------------------------------
// out[b,c] = sum_hw Wt[b,hw] * cnn[b,c,hw]   (used for ctx and masked avg)
// one wave per (b,c)
// ---------------------------------------------------------------------------
__global__ void wsum_kernel(const float* __restrict__ Wt, const float* __restrict__ cnn,
                            float* __restrict__ out, int C)
{
    const int idx  = blockIdx.x * 8 + (threadIdx.x >> 5);
    const int lane = threadIdx.x & 31;
    if (idx >= B_ * C) return;
    const int b = idx / C, c = idx % C;
    const float* row = cnn + ((size_t)b * C + c) * HW_;
    const float* wv  = Wt + (size_t)b * HW_;
    float acc = 0.f;
    for (int k = lane; k < HW_; k += 32) acc += row[k] * wv[k];
    for (int off = 16; off > 0; off >>= 1) acc += __shfl_down(acc, off, 32);
    if (lane == 0) out[(size_t)b * C + c] = acc;
}

// ---------------------------------------------------------------------------
// Gather embedded tokens for all steps:  WE[t*16+b, :] = emb[tok(t,b), :]
// tok(0,b)=1 ; tok(t,b)=labels[b, t-1]
// ---------------------------------------------------------------------------
__global__ void gather_emb_kernel(const int* __restrict__ labels,
                                  const float* __restrict__ emb,
                                  float* __restrict__ WE)
{
    const int m = blockIdx.x;         // 0..575
    const int k = threadIdx.x;        // 0..255
    const int t = m >> 4, b = m & 15;
    const int tok = (t == 0) ? 1 : labels[b * T_ + (t - 1)];
    WE[(size_t)m * HID_ + k] = emb[(size_t)tok * HID_ + k];
}

__global__ void bcomb_kernel(const float* a, const float* b, const float* c, float* o)
{ const int i = threadIdx.x; o[i] = a[i] + b[i] + c[i]; }

__global__ void zero_kernel(float* p, int n)
{ const int i = blockIdx.x * blockDim.x + threadIdx.x; if (i < n) p[i] = 0.f; }

// ---------------------------------------------------------------------------
// GRU gate elementwise: hidden <- (1-z)*n + z*hidden
// ---------------------------------------------------------------------------
__global__ void gru_kernel(const float* __restrict__ GIt, const float* __restrict__ GH,
                           float* __restrict__ hidden)
{
    const int i = blockIdx.x * blockDim.x + threadIdx.x;   // 0..4095
    const int b = i >> 8, n = i & 255;
    const float ir  = GIt[b * 768 + n];
    const float iz  = GIt[b * 768 + 256 + n];
    const float inn = GIt[b * 768 + 512 + n];
    const float hr  = GH[b * 768 + n];
    const float hz  = GH[b * 768 + 256 + n];
    const float hn  = GH[b * 768 + 512 + n];
    const float r = 1.0f / (1.0f + expf(-(ir + hr)));
    const float z = 1.0f / (1.0f + expf(-(iz + hz)));
    const float nst = tanhf(inn + r * hn);
    const float h = hidden[i];
    hidden[i] = (1.0f - z) * nst + z * h;
}

// ---------------------------------------------------------------------------
// Fused coverage-conv + score + energy:
//   cov[b,hw,a] = sum_tap MfT[tap,a] * alpha_sum_pad[b, hw+tap]
//   energy[b,hw] = sum_a tanh(Q[b,a] + TR[b,hw,a] + cov) * acw[a] + acb
// One wave per (b,hw); lane handles a = lane + 32*j, j=0..15.
// MfT is [tap][a] so the 16 loads per tap are lane-coalesced.
// ---------------------------------------------------------------------------
__global__ void energy_kernel(const float* __restrict__ Q, const float* __restrict__ TR,
                              const float* __restrict__ MfT, const float* __restrict__ asum,
                              const float* __restrict__ acw, const float* __restrict__ acb,
                              float* __restrict__ E)
{
    const int wave = threadIdx.x >> 5, lane = threadIdx.x & 31;
    const int hw = blockIdx.x * 8 + wave;
    const int b  = blockIdx.y;
    const int h = hw >> 5, w = hw & 31;
    const float* asb = asum + (size_t)b * HW_;

    float cov[16];
#pragma unroll
    for (int j = 0; j < 16; ++j) cov[j] = 0.f;

    for (int tap = 0; tap < 121; ++tap) {
        const int dy = tap / 11 - 5, dx = tap % 11 - 5;
        const int hh = h + dy, ww = w + dx;
        const float av = (hh >= 0 && hh < H_ && ww >= 0 && ww < W_) ? asb[hh * 32 + ww] : 0.f;
        const float* mrow = MfT + (size_t)tap * ATT_ + lane;
#pragma unroll
        for (int j = 0; j < 16; ++j)
            cov[j] += mrow[j << 5] * av;
    }

    const float* qb  = Q + (size_t)b * ATT_;
    const float* trb = TR + ((size_t)b * HW_ + hw) * ATT_;
    float acc = 0.f;
#pragma unroll
    for (int j = 0; j < 16; ++j) {
        const int a = lane + (j << 5);
        const float s = tanhf(qb[a] + trb[a] + cov[j]);
        acc += s * acw[a];
    }
    for (int off = 16; off > 0; off >>= 1) acc += __shfl_down(acc, off, 32);
    if (lane == 0) E[(size_t)b * HW_ + hw] = acc + acb[0];
}

// ---------------------------------------------------------------------------
// Per-batch softmax over hw (global max subtraction cancels -> per-b max),
// alpha_sum accumulation, and write alphas output slab for step t.
// one block per b, 512 threads
// ---------------------------------------------------------------------------
__global__ void softmax_kernel(const float* __restrict__ E, const float* __restrict__ mask2d,
                               float* __restrict__ alpha, float* __restrict__ asum,
                               float* __restrict__ alphas_out, int t)
{
    const int b = blockIdx.x, tid = threadIdx.x;
    __shared__ float red[HW_];
    const float e = E[b * HW_ + tid];
    red[tid] = e;
    __syncthreads();
    for (int s = 256; s > 0; s >>= 1) { if (tid < s) red[tid] = fmaxf(red[tid], red[tid + s]); __syncthreads(); }
    const float mx = red[0];
    __syncthreads();
    const float ex = expf(e - mx) * mask2d[b * HW_ + tid];
    red[tid] = ex;
    __syncthreads();
    for (int s = 256; s > 0; s >>= 1) { if (tid < s) red[tid] += red[tid + s]; __syncthreads(); }
    const float a = ex / (red[0] + 1e-10f);
    alpha[b * HW_ + tid] = a;
    asum[b * HW_ + tid] += a;
    alphas_out[(size_t)b * T_ * HW_ + (size_t)t * HW_ + tid] = a;
}

// ---------------------------------------------------------------------------
// host side
// ---------------------------------------------------------------------------
static inline void gemm(hipStream_t s,
                        const float* A, int rsA, int csA, int batA,
                        const float* Bw, int rsB, int csB,
                        const float* bias,
                        const float* src, int rsS, int srcMmod,
                        float* D, int rsD, int csD, int batD,
                        int M, int N, int K, int batch, int act)
{
    dim3 g((N + 15) / 16, (M + 15) / 16, batch);
    wmma_gemm_kernel<<<g, dim3(32), 0, s>>>(A, rsA, csA, batA, Bw, rsB, csB, bias,
                                            src, rsS, srcMmod, D, rsD, csD, batD,
                                            M, N, K, act);
}

extern "C" void kernel_launch(void* const* d_in, const int* in_sizes, int n_in,
                              void* d_out, int out_size, void* d_ws, size_t ws_size,
                              hipStream_t stream)
{
    const float* cnn       = (const float*)d_in[0];
    const int*   labels    = (const int*)  d_in[1];
    const float* cnt_preds = (const float*)d_in[2];
    const float* imask     = (const float*)d_in[3];
    // d_in[4] labels_mask: unused in forward outputs
    const float* init_w    = (const float*)d_in[5];
    const float* init_b    = (const float*)d_in[6];
    const float* emb       = (const float*)d_in[7];
    const float* gru_wih   = (const float*)d_in[8];
    const float* gru_whh   = (const float*)d_in[9];
    const float* gru_bih   = (const float*)d_in[10];
    const float* gru_bhh   = (const float*)d_in[11];
    const float* att_hw_w  = (const float*)d_in[12];
    const float* att_hw_b  = (const float*)d_in[13];
    const float* att_conv_w= (const float*)d_in[14];
    const float* att_ww_w  = (const float*)d_in[15];
    const float* att_ac_w  = (const float*)d_in[16];
    const float* att_ac_b  = (const float*)d_in[17];
    const float* enc_conv_w= (const float*)d_in[18];
    const float* enc_conv_b= (const float*)d_in[19];
    const float* ws_w      = (const float*)d_in[20];
    const float* ws_b      = (const float*)d_in[21];
    const float* wemb_w    = (const float*)d_in[22];
    const float* wemb_b    = (const float*)d_in[23];
    const float* wctx_w    = (const float*)d_in[24];
    const float* wctx_b    = (const float*)d_in[25];
    const float* cc_w      = (const float*)d_in[26];
    const float* cc_b      = (const float*)d_in[27];
    const float* wconv_w   = (const float*)d_in[28];
    const float* wconv_b   = (const float*)d_in[29];

    float* outp   = (float*)d_out;                    // probs (B,T,WN)
    float* outa   = outp + (size_t)B_ * T_ * WN_;     // alphas (B,T,H,W)

    // workspace layout (floats)
    float* w = (float*)d_ws;
    size_t o = 0;
    auto alloc = [&](size_t n) { float* p = w + o; o += (n + 63) & ~(size_t)63; return p; };
    float* WE   = alloc((size_t)T_ * B_ * HID_);      // 576x256 gathered embeddings
    float* GI   = alloc((size_t)T_ * B_ * 768);       // all-step GRU input gates
    float* OB   = alloc((size_t)T_ * B_ * HID_);      // out-base per step
    float* CCTX = alloc((size_t)B_ * HID_);           // counting ctx
    float* BC   = alloc(HID_);                        // combined biases
    float* MFT  = alloc((size_t)128 * ATT_);          // fused coverage weights [tap][a]
    float* TR   = alloc((size_t)B_ * HW_ * ATT_);     // trans_hwc (+pos)
    float* M2D  = alloc((size_t)B_ * HW_);
    float* YN   = alloc((size_t)B_ * HW_);
    float* XN   = alloc((size_t)B_ * HW_);
    float* W0   = alloc((size_t)B_ * HW_);
    float* AVG  = alloc((size_t)B_ * C_);
    float* HID  = alloc((size_t)B_ * HID_);
    float* GH   = alloc((size_t)B_ * 768);
    float* Q    = alloc((size_t)B_ * ATT_);
    float* E    = alloc((size_t)B_ * HW_);
    float* AL   = alloc((size_t)B_ * HW_);
    float* AS   = alloc((size_t)B_ * HW_);
    float* CTX  = alloc((size_t)B_ * C_);
    float* OV   = alloc((size_t)B_ * HID_);
    (void)ws_size; (void)in_sizes; (void)n_in; (void)out_size;

    // ---- one-time precompute ----
    maskcum_kernel<<<dim3(B_), dim3(HW_), 0, stream>>>(imask, M2D, YN, XN, W0);
    // masked average pooling -> hidden0
    wsum_kernel<<<dim3((B_ * C_ + 7) / 8), dim3(256), 0, stream>>>(W0, cnn, AVG, C_);
    gemm(stream, AVG, C_, 1, 0, init_w, C_, 1, init_b, nullptr, 0, 0,
         HID, HID_, 1, 0, B_, HID_, C_, 1, /*tanh*/1);
    zero_kernel<<<dim3((B_ * HW_ + 255) / 256), dim3(256), 0, stream>>>(AS, B_ * HW_);
    bcomb_kernel<<<dim3(1), dim3(HID_), 0, stream>>>(ws_b, wemb_b, wctx_b, BC);
    gather_emb_kernel<<<dim3(T_ * B_), dim3(HID_), 0, stream>>>(labels, emb, WE);
    // counting ctx = preds @ cc_w^T + cc_b
    gemm(stream, cnt_preds, CNTN_, 1, 0, cc_w, CNTN_, 1, cc_b, nullptr, 0, 0,
         CCTX, HID_, 1, 0, B_, HID_, CNTN_, 1, 0);
    // OB[t*16+b] = we @ wemb^T + (ws_b+wemb_b+wctx_b) + counting_ctx[b]
    gemm(stream, WE, HID_, 1, 0, wemb_w, HID_, 1, BC, CCTX, HID_, B_,
         OB, HID_, 1, 0, T_ * B_, HID_, HID_, 1, 0);
    // GI = we @ wih^T + bih for all steps (carry-independent, hoisted)
    gemm(stream, WE, HID_, 1, 0, gru_wih, HID_, 1, gru_bih, nullptr, 0, 0,
         GI, 768, 1, 0, T_ * B_, 768, HID_, 1, 0);
    // MfT[tap,a] = sum_c ww[a,c] * conv_w[c,0,tap]  (conv/projection fusion,
    // stored transposed for lane-coalesced reads in energy_kernel)
    gemm(stream, att_ww_w, ATT_, 1, 0, att_conv_w, 1, 121, nullptr, nullptr, 0, 0,
         MFT, 1, ATT_, 0, ATT_, 121, ATT_, 1, 0);
    // trans_hwc[b,hw,a] = cnn[b,:,hw] @ enc_w[a,:] + enc_b  (batched over b)
    gemm(stream, cnn, 1, HW_, C_ * HW_, enc_conv_w, C_, 1, enc_conv_b,
         nullptr, 0, 0, TR, ATT_, 1, HW_ * ATT_, HW_, ATT_, C_, B_, 0);
    posadd_kernel<<<dim3(HW_, B_), dim3(ATT_), 0, stream>>>(YN, XN, TR);

    // ---- sequential scan over T steps ----
    for (int t = 0; t < T_; ++t) {
        // GRU hidden gates + update
        gemm(stream, HID, HID_, 1, 0, gru_whh, HID_, 1, gru_bhh, nullptr, 0, 0,
             GH, 768, 1, 0, B_, 768, HID_, 1, 0);
        gru_kernel<<<dim3(16), dim3(256), 0, stream>>>(GI + (size_t)t * B_ * 768, GH, HID);
        // attention query
        gemm(stream, HID, HID_, 1, 0, att_hw_w, HID_, 1, att_hw_b, nullptr, 0, 0,
             Q, ATT_, 1, 0, B_, ATT_, HID_, 1, 0);
        // fused coverage conv + tanh + energy reduce
        energy_kernel<<<dim3(HW_ / 8, B_), dim3(256), 0, stream>>>(Q, TR, MFT, AS,
                                                                   att_ac_w, att_ac_b, E);
        // softmax + alpha_sum update + alphas output
        softmax_kernel<<<dim3(B_), dim3(HW_), 0, stream>>>(E, M2D, AL, AS, outa, t);
        // context vector
        wsum_kernel<<<dim3((B_ * C_ + 7) / 8), dim3(256), 0, stream>>>(AL, cnn, CTX, C_);
        // out = hidden@ws^T + OB[t]  ;  out += ctx@wctx^T  ;  prob = out@wconv^T + b
        gemm(stream, HID, HID_, 1, 0, ws_w, HID_, 1, nullptr,
             OB + (size_t)t * B_ * HID_, HID_, 0, OV, HID_, 1, 0, B_, HID_, HID_, 1, 0);
        gemm(stream, CTX, C_, 1, 0, wctx_w, C_, 1, nullptr, OV, HID_, 0,
             OV, HID_, 1, 0, B_, HID_, C_, 1, 0);
        gemm(stream, OV, HID_, 1, 0, wconv_w, HID_, 1, wconv_b, nullptr, 0, 0,
             outp + (size_t)t * WN_, T_ * WN_, 1, 0, B_, WN_, HID_, 1, 0);
    }
}
